// GCN_60155311947854
// MI455X (gfx1250) — matrix operand order repaired and checked by reading, not compiled, not run on hardware
//
#include <hip/hip_runtime.h>

#define DD   128
#define RESC 0.2f
#define EPSC 1e-5f

typedef __attribute__((ext_vector_type(16))) __bf16 v16bf;
typedef __attribute__((ext_vector_type(8)))  float  v8f;

// ---------------------------------------------------------------------------
// utility kernels
// ---------------------------------------------------------------------------
__global__ void k_zero_f32(float* __restrict__ p, long n) {
    long i = (long)blockIdx.x * blockDim.x + threadIdx.x;
    if (i < n) p[i] = 0.0f;
}

__global__ void k_degree(const int* __restrict__ src, const int* __restrict__ dst,
                         float* __restrict__ degO, float* __restrict__ degI, int E) {
    int e = blockIdx.x * blockDim.x + threadIdx.x;
    if (e >= E) return;
    atomicAdd(&degO[src[e]], 1.0f);
    atomicAdd(&degI[dst[e]], 1.0f);
}

__global__ void k_rsqrt_deg(float* __restrict__ d, int n) {
    int i = blockIdx.x * blockDim.x + threadIdx.x;
    if (i < n) d[i] = rsqrtf(fmaxf(d[i], 1.0f));
}

// Pack W (f32 row-major [K][N], 128x128) into per-lane WMMA B-tile layout:
// slot = ((kt*8 + nt)*32 + lane), 16 consecutive bf16 per slot.
// For lane L: column n = nt*16 + (L&15); K = kt*32 + (L<16 ? 0 : 16) + r, r=0..15.
__global__ void k_packW(const float* __restrict__ W, __bf16* __restrict__ Wp) {
    int t = blockIdx.x * blockDim.x + threadIdx.x;   // 0 .. 16383
    if (t >= DD * DD) return;
    int r    = t & 15;
    int slot = t >> 4;
    int L    = slot & 31;
    int nt   = (slot >> 5) & 7;
    int kt   = slot >> 8;
    int n = nt * 16 + (L & 15);
    int K = kt * 32 + ((L < 16) ? 0 : 16) + r;
    Wp[t] = (__bf16)W[K * DD + n];
}

// xbf[row][c] = bf16( x[row][c] * rout[row] )
__global__ void k_scale_in(const float* __restrict__ x, const float* __restrict__ rout,
                           __bf16* __restrict__ xbf, long total) {
    long i = (long)blockIdx.x * blockDim.x + threadIdx.x;
    if (i >= total) return;
    int row = (int)(i >> 7);
    xbf[i] = (__bf16)(x[i] * rout[row]);
}

// ---------------------------------------------------------------------------
// GEMM: C[N,128] = A_bf16[N,128] @ Wp(bf16 packed 128x128), f32 accumulate.
// Block = 8 waves. The 32KB packed W is staged into LDS ONCE per block
// (one global fetch per block instead of per wave), then each wave feeds its
// 32 WMMAs from ds_load_b128 (short-latency WGP-local path), keeping the vmem
// pipe free for A loads / C stores.
// ---------------------------------------------------------------------------
__global__ void __launch_bounds__(256)
k_gemm(const __bf16* __restrict__ A, const __bf16* __restrict__ Wp,
       float* __restrict__ C, int mtiles, int Nrows) {
    __shared__ __bf16 sw[DD * DD];           // 32 KB of 320 KB/WGP

    // cooperative staging: 2048 x int4, 8 per thread
    {
        const int4* s4 = (const int4*)Wp;
        int4*       d4 = (int4*)sw;
#pragma unroll
        for (int i = 0; i < (DD * DD / 8) / 256; ++i)
            d4[threadIdx.x + i * 256] = s4[threadIdx.x + i * 256];
    }
    __syncthreads();

    int lane = threadIdx.x & 31;
    int wave = threadIdx.x >> 5;
    int mt   = blockIdx.x * 8 + wave;
    if (mt >= mtiles) return;                // safe: after the only barrier

    int mrow = (mt << 4) + (lane & 15);      // A row this lane supplies
    if (mrow >= Nrows) mrow = Nrows - 1;     // clamp (EXEC must stay full)
    int lo = (lane < 16) ? 0 : 16;
    const __bf16* arow = A + (long)mrow * DD;

    union U { v16bf v; int4 q[2]; };

    // Stage all four A K-slices (16 bf16 each) up front: 8 x global_load_b128.
    U a[4];
#pragma unroll
    for (int kt = 0; kt < 4; ++kt) {
        const __bf16* ap = arow + kt * 32 + lo;
        a[kt].q[0] = *(const int4*)(ap);
        a[kt].q[1] = *(const int4*)(ap + 8);
    }

    v8f zero = {0.f, 0.f, 0.f, 0.f, 0.f, 0.f, 0.f, 0.f};
    v8f acc[8];
#pragma unroll
    for (int i = 0; i < 8; ++i) acc[i] = zero;

    const __bf16* wlane = sw + (lane << 4);

#pragma unroll
    for (int kt = 0; kt < 4; ++kt) {
#pragma unroll
        for (int nt = 0; nt < 8; ++nt) {
            U b;
            const __bf16* bp = wlane + (((kt * 8 + nt) * 32) << 4);
            b.q[0] = *(const int4*)(bp);         // ds_load_b128
            b.q[1] = *(const int4*)(bp + 8);     // ds_load_b128
            acc[nt] = __builtin_amdgcn_wmma_f32_16x16x32_bf16(
                false, a[kt].v, false, b.v, (short)0, acc[nt], false, false);
        }
    }

    // C/D layout: VGPR j -> row = mt*16 + j + (lane<16?0:8), col = nt*16 + (lane&15)
    int rbase   = (mt << 4) + ((lane < 16) ? 0 : 8);
    int colbase = lane & 15;
#pragma unroll
    for (int nt = 0; nt < 8; ++nt) {
        union { v8f v; float f[8]; } cu;
        cu.v = acc[nt];
        int col = nt * 16 + colbase;
#pragma unroll
        for (int j = 0; j < 8; ++j) {
            int rr = rbase + j;
            if (rr < Nrows) C[(long)rr * DD + col] = cu.f[j];
        }
    }
}

// ---------------------------------------------------------------------------
// scatter-add: agg[dst[e]] += xw[src[e]]   (32 lanes/edge, 4 floats/lane)
// xw and agg both fit in the 192MB L2 -> atomics mostly resolve in L2.
// ---------------------------------------------------------------------------
__global__ void k_scatter(const int* __restrict__ src, const int* __restrict__ dst,
                          const float* __restrict__ xw, float* __restrict__ agg, int E) {
    long tid = (long)blockIdx.x * blockDim.x + threadIdx.x;
    int e = (int)(tid >> 5);
    if (e >= E) return;
    int c0 = (int)(tid & 31) << 2;
    const float4 v = *(const float4*)(xw + (long)src[e] * DD + c0);
    float* a = agg + (long)dst[e] * DD + c0;
    atomicAdd(a + 0, v.x);
    atomicAdd(a + 1, v.y);
    atomicAdd(a + 2, v.z);
    atomicAdd(a + 3, v.w);
}

// ---------------------------------------------------------------------------
// fused epilogue: h = agg*rin + bias + RES*in ; LN(g,b) ; relu ; *rout ; ->bf16
// one 128-thread block per row
// ---------------------------------------------------------------------------
__global__ void k_ln_epilogue(const float* __restrict__ agg, const float* __restrict__ rin,
                              const float* __restrict__ rout, const float* __restrict__ bias,
                              const float* __restrict__ infeat,
                              const float* __restrict__ g, const float* __restrict__ bb,
                              __bf16* __restrict__ xout) {
    __shared__ float red[DD];
    int row = blockIdx.x;
    int c   = threadIdx.x;
    long idx = (long)row * DD + c;

    float t = agg[idx] * rin[row] + bias[c] + RESC * infeat[idx];

    red[c] = t;
    __syncthreads();
#pragma unroll
    for (int s = 64; s > 0; s >>= 1) { if (c < s) red[c] += red[c + s]; __syncthreads(); }
    float mu = red[0] * (1.0f / DD);
    __syncthreads();

    float dv = t - mu;
    red[c] = dv * dv;
    __syncthreads();
#pragma unroll
    for (int s = 64; s > 0; s >>= 1) { if (c < s) red[c] += red[c + s]; __syncthreads(); }
    float var = red[0] * (1.0f / DD);

    float y = dv * rsqrtf(var + EPSC) * g[c] + bb[c];
    y = fmaxf(y, 0.0f);
    xout[idx] = (__bf16)(y * rout[row]);
}

// final: out = agg*rin + bias + RES*in   (f32)
__global__ void k_final(const float* __restrict__ agg, const float* __restrict__ rin,
                        const float* __restrict__ bias, const float* __restrict__ infeat,
                        float* __restrict__ out, long total) {
    long i = (long)blockIdx.x * blockDim.x + threadIdx.x;
    if (i >= total) return;
    int row = (int)(i >> 7);
    int c   = (int)(i & (DD - 1));
    out[i] = agg[i] * rin[row] + bias[c] + RESC * infeat[i];
}

// ---------------------------------------------------------------------------
extern "C" void kernel_launch(void* const* d_in, const int* in_sizes, int n_in,
                              void* d_out, int out_size, void* d_ws, size_t ws_size,
                              hipStream_t stream) {
    const int*   src     = (const int*)d_in[0];
    const int*   dst     = (const int*)d_in[1];
    const float* in_feat = (const float*)d_in[2];
    const float* W1      = (const float*)d_in[3];
    const float* b1      = (const float*)d_in[4];
    const float* W2      = (const float*)d_in[5];
    const float* b2      = (const float*)d_in[6];
    const float* g1      = (const float*)d_in[7];
    const float* bb1     = (const float*)d_in[8];
    const float* g2      = (const float*)d_in[9];
    const float* bb2     = (const float*)d_in[10];
    float* out = (float*)d_out;

    const int E = in_sizes[0];
    const int N = in_sizes[2] / DD;
    const long nd = (long)N * DD;

    // workspace layout
    char* ws = (char*)d_ws;
    size_t off = 0;
    auto take = [&](size_t bytes) { void* p = ws + off; off += (bytes + 255) & ~(size_t)255; return p; };
    float*  xw   = (float*)take(nd * sizeof(float));
    float*  agg  = (float*)take(nd * sizeof(float));
    __bf16* xbf  = (__bf16*)take(nd * sizeof(__bf16));
    float*  rout = (float*)take((size_t)N * sizeof(float));
    float*  rin  = (float*)take((size_t)N * sizeof(float));
    __bf16* W1p  = (__bf16*)take((size_t)DD * DD * sizeof(__bf16));
    __bf16* W2p  = (__bf16*)take((size_t)DD * DD * sizeof(__bf16));

    const int B = 256;
    const int mtiles = (N + 15) / 16;
    const int gemmBlocks = (mtiles + 7) / 8;
    const long scatterThreads = (long)E * 32;

    // degrees -> rsqrt
    k_zero_f32<<<(N + B - 1) / B, B, 0, stream>>>(rout, N);
    k_zero_f32<<<(N + B - 1) / B, B, 0, stream>>>(rin, N);
    k_degree<<<(E + B - 1) / B, B, 0, stream>>>(src, dst, rout, rin, E);
    k_rsqrt_deg<<<(N + B - 1) / B, B, 0, stream>>>(rout, N);
    k_rsqrt_deg<<<(N + B - 1) / B, B, 0, stream>>>(rin, N);

    // pack weights to WMMA-friendly bf16 layout
    k_packW<<<(DD * DD + B - 1) / B, B, 0, stream>>>(W1, W1p);
    k_packW<<<(DD * DD + B - 1) / B, B, 0, stream>>>(W2, W2p);

    // conv1 input: bf16(in_feat * rout)
    k_scale_in<<<(int)((nd + B - 1) / B), B, 0, stream>>>(in_feat, rout, xbf, nd);

    // ---- conv1 ----
    k_zero_f32<<<(int)((nd + B - 1) / B), B, 0, stream>>>(agg, nd);
    k_gemm<<<gemmBlocks, B, 0, stream>>>(xbf, W1p, xw, mtiles, N);
    k_scatter<<<(int)((scatterThreads + B - 1) / B), B, 0, stream>>>(src, dst, xw, agg, E);
    k_ln_epilogue<<<N, DD, 0, stream>>>(agg, rin, rout, b1, in_feat, g1, bb1, xbf);

    // ---- conv2 ----
    k_zero_f32<<<(int)((nd + B - 1) / B), B, 0, stream>>>(agg, nd);
    k_gemm<<<gemmBlocks, B, 0, stream>>>(xbf, W2p, xw, mtiles, N);
    k_scatter<<<(int)((scatterThreads + B - 1) / B), B, 0, stream>>>(src, dst, xw, agg, E);
    k_ln_epilogue<<<N, DD, 0, stream>>>(agg, rin, rout, b2, in_feat, g2, bb2, xbf);

    // ---- conv3 ----
    k_zero_f32<<<(int)((nd + B - 1) / B), B, 0, stream>>>(agg, nd);
    k_gemm<<<gemmBlocks, B, 0, stream>>>(xbf, W2p, xw, mtiles, N);
    k_scatter<<<(int)((scatterThreads + B - 1) / B), B, 0, stream>>>(src, dst, xw, agg, E);
    k_final<<<(int)((nd + B - 1) / B), B, 0, stream>>>(agg, rin, b2, in_feat, out, nd);
}
